// BronzeAgeGNN_90134183674239
// MI455X (gfx1250) — matrix operand order repaired and checked by reading, not compiled, not run on hardware
//
#include <hip/hip_runtime.h>

#define NN   50000
#define DD   128
#define EE   800000
#define BB   10
#define LLAY 4
#define OUTC 40
#define KTOT (DD * (BB + 1))   // 1408
#define MT   32                // nodes per block in the layer GEMM

typedef __attribute__((ext_vector_type(16))) _Float16 v16h;
typedef __attribute__((ext_vector_type(8)))  _Float16 v8h;
typedef __attribute__((ext_vector_type(8)))  float    v8f;

// gfx1250 async global->LDS bulk copy (ASYNCcnt-tracked), guarded so the
// build cannot regress on toolchains that lack the builtins.
#if defined(__has_builtin)
#if __has_builtin(__builtin_amdgcn_global_load_async_to_lds_b128) && \
    __has_builtin(__builtin_amdgcn_s_wait_asynccnt)
#define USE_ASYNC_COPY 1
#endif
#endif
#ifndef USE_ASYNC_COPY
#define USE_ASYNC_COPY 0
#endif

// Builtin signature (from hipcc diagnostic): (v4i AS1*, v4i AS3*, imm, imm)
typedef int v4i_vs __attribute__((__vector_size__(16)));
typedef __attribute__((address_space(1))) v4i_vs* gbl_v4i_ptr;
typedef __attribute__((address_space(3))) v4i_vs* lds_v4i_ptr;

// ---------------------------------------------------------------------------
// WMMA fragment loaders (wave32, V_WMMA_F32_16X16X32_F16)
// A (16x32 f16): lanes 0-15 row M=lane, K {k0..k0+15};
//                lanes 16-31 row M=lane-16, K {k0+16..k0+31}
// B (32x16 f16): lanes 0-15 col N=lane, K {k0..k0+15};
//                lanes 16-31 col N=lane-16, K {k0+16..k0+31}
// Both read from row-major [row][ldk] f16 arrays.
// ---------------------------------------------------------------------------
__device__ __forceinline__ v16h load_a_frag(const _Float16* base, int ldk, int k0, int lane) {
  int row = lane & 15;
  int kb  = k0 + ((lane >> 4) << 4);
  return *(const v16h*)(base + row * ldk + kb);
}

__device__ __forceinline__ v16h load_b_frag(const _Float16* base, int ldk, int k0, int lane) {
  int col = lane & 15;
  int kb  = k0 + ((lane >> 4) << 4);
  return *(const v16h*)(base + col * ldk + kb);
}

#define WMMA_F16(acc, a, b) \
  __builtin_amdgcn_wmma_f32_16x16x32_f16(false, (a), false, (b), (short)0, (acc), false, false)

// ---------------------------------------------------------------------------
// Small utility kernels
// ---------------------------------------------------------------------------
__global__ void f32_to_f16_kernel(const float* __restrict__ in, _Float16* __restrict__ out, int n) {
  int i = blockIdx.x * blockDim.x + threadIdx.x;
  if (i < n) out[i] = (_Float16)in[i];
}

// W [rows x cols] f32 -> WT [cols x rows] f16
__global__ void transpose_f16_kernel(const float* __restrict__ W, _Float16* __restrict__ WT,
                                     int rows, int cols) {
  int i = blockIdx.x * blockDim.x + threadIdx.x;
  if (i < rows * cols) {
    int j = i / rows;
    int k = i - j * rows;
    WT[i] = (_Float16)W[(size_t)k * cols + j];
  }
}

__global__ void zero_u32_kernel(unsigned* __restrict__ p, int n) {
  int i = blockIdx.x * blockDim.x + threadIdx.x;
  if (i < n) p[i] = 0u;
}

// segment_sum of one-hot rows == integer histogram per (dst, argmax(src))
__global__ void edge_count_kernel(const long long* __restrict__ src,
                                  const long long* __restrict__ dst,
                                  const int* __restrict__ idx,
                                  unsigned* __restrict__ cnt) {
  int e = blockIdx.x * blockDim.x + threadIdx.x;
  if (e < EE) {
    int s = (int)src[e];
    int d = (int)dst[e];
    atomicAdd(&cnt[(size_t)d * DD + idx[s]], 1u);
  }
}

// ---------------------------------------------------------------------------
// Input layer:  argmax(x @ W_in + b_in)  via WMMA.  16 nodes / block, 8 waves,
// each wave owns one 16-wide output tile; K = 128 (4 wmma steps).
// ---------------------------------------------------------------------------
__global__ __launch_bounds__(256)
void gnn_input_kernel(const _Float16* __restrict__ xh,      // [N][128] f16
                      const _Float16* __restrict__ WTin,    // [128][128] f16 (transposed)
                      const float* __restrict__ bias,       // [128]
                      int* __restrict__ idx_out) {
  __shared__ __align__(64) float sLogit[16 * DD];

  const int tid  = threadIdx.x;
  const int n0   = blockIdx.x * 16;
  const int wave = tid >> 5;
  const int lane = tid & 31;
  const int j0   = wave * 16;

  const _Float16* abase = xh + (size_t)n0 * DD;
  const _Float16* wbase = WTin + (size_t)j0 * DD;

  v8f acc = {};
  #pragma unroll
  for (int k0 = 0; k0 < DD; k0 += 32) {
    v16h a = load_a_frag(abase, DD, k0, lane);
    v16h b = load_b_frag(wbase, DD, k0, lane);
    acc = WMMA_F16(acc, a, b);
  }

  const int mbase = (lane < 16) ? 0 : 8;
  const int ncol  = j0 + (lane & 15);
  #pragma unroll
  for (int r = 0; r < 8; ++r)
    sLogit[(mbase + r) * DD + ncol] = acc[r];
  __syncthreads();

  if (tid < 16) {
    const float* lrow = sLogit + tid * DD;
    float best = -1e30f; int bj = 0;
    for (int j = 0; j < DD; ++j) {
      float v = lrow[j] + bias[j];
      if (v > best) { best = v; bj = j; }
    }
    idx_out[n0 + tid] = bj;
  }
}

// ---------------------------------------------------------------------------
// GNN layer:  argmax([onehot(idx) | states(cnt)] @ W_l + b_l)
//  - 32-node M-tile per block (halves L2 weight traffic vs 16-node tiles)
//  - counts staged into LDS with global_load_async_to_lds_b128 (ASYNCcnt)
//  - A-tile [32 x 1408] f16 in LDS (one-hot + 110-entry sigmoid/elu LUT)
//  - 8 waves, one 16-col tile each; each B fragment feeds TWO v_wmma ops
//  - counts region and logits region share one 16 KB LDS block (disjoint
//    lifetimes separated by a barrier)
// ---------------------------------------------------------------------------
__global__ __launch_bounds__(256)
void gnn_layer_kernel(const unsigned* __restrict__ cnt,     // [N][128]
                      const int* __restrict__ idx_in,       // [N]
                      const _Float16* __restrict__ WT,      // [128][1408] f16 (transposed)
                      const float* __restrict__ bias,       // [128]
                      int* __restrict__ idx_out) {
  __shared__ __align__(64) _Float16 sA[MT * KTOT];          // 88 KB
  __shared__ __align__(64) unsigned sU[MT * DD];            // 16 KB: counts, then logits
  __shared__ float sLut[112];

  const int tid = threadIdx.x;
  const int n0  = blockIdx.x * MT;

  // states LUT: f(c,y) = sigmoid(10*(elu(c-y) - 0.5)), c in 0..10, y in 0..9
  if (tid < (BB + 1) * BB) {
    int c = tid / BB, y = tid - (tid / BB) * BB;
    float z = (float)(c - y);
    float e = z > 0.0f ? z : (__expf(z) - 1.0f);
    sLut[tid] = 1.0f / (1.0f + __expf(-10.0f * (e - 0.5f)));
  }

  // Stage the 32x128 u32 count tile into LDS (16 KB, contiguous).
#if USE_ASYNC_COPY
  {
    const char* gsrc = (const char*)(cnt + (size_t)n0 * DD);
    char*       ldst = (char*)sU;
    #pragma unroll
    for (int i = 0; i < (MT * DD * 4) / (256 * 16); ++i) {   // 4 x b128 per thread
      int off = (tid + i * 256) * 16;
      __builtin_amdgcn_global_load_async_to_lds_b128(
          (gbl_v4i_ptr)(unsigned long long)(gsrc + off),
          (lds_v4i_ptr)(unsigned long long)(ldst + off), 0, 0);
    }
    __builtin_amdgcn_s_wait_asynccnt(0);
  }
#else
  for (int i = tid; i < MT * DD; i += 256) {
    size_t g = (size_t)n0 * DD + i;
    if (g >= (size_t)NN * DD) g = (size_t)NN * DD - 1;
    sU[i] = cnt[g];
  }
#endif
  __syncthreads();

  // Build A tile: 32 rows x 1408 K.  8 threads per row, 176 K-values each.
  {
    const int row  = tid >> 3;
    const int tsub = tid & 7;
    int node = n0 + row; if (node >= NN) node = NN - 1;
    const int hidx = idx_in[node];
    const unsigned* crow = sU + row * DD;
    _Float16* arow = sA + row * KTOT;
    const int kbeg = tsub * (KTOT / 8);
    for (int k = kbeg; k < kbeg + (KTOT / 8); ++k) {
      float v;
      if (k < DD) {
        v = (k == hidx) ? 1.0f : 0.0f;
      } else {
        int kk = k - DD;
        int d  = kk / BB;
        int y  = kk - d * BB;
        unsigned c = crow[d];
        if (c > (unsigned)BB) c = (unsigned)BB;
        v = sLut[c * BB + y];
      }
      arow[k] = (_Float16)v;
    }
  }
  __syncthreads();           // counts fully consumed; sU becomes logits below

  const int wave = tid >> 5;
  const int lane = tid & 31;
  const int j0   = wave * 16;
  const _Float16* wbase = WT + (size_t)j0 * KTOT;

  v8f acc0 = {}, acc1 = {};
  for (int k0 = 0; k0 < KTOT; k0 += 32) {
    __builtin_prefetch((const void*)(wbase + (lane & 15) * KTOT + k0 + 32), 0, 1);
    v16h b  = load_b_frag(wbase, KTOT, k0, lane);
    v16h a0 = load_a_frag(sA, KTOT, k0, lane);
    v16h a1 = load_a_frag(sA + 16 * KTOT, KTOT, k0, lane);
    acc0 = WMMA_F16(acc0, a0, b);          // rows 0..15
    acc1 = WMMA_F16(acc1, a1, b);          // rows 16..31 (B fragment reused)
  }

  float* sLogit = (float*)sU;              // reuse the 16 KB region
  const int mbase = (lane < 16) ? 0 : 8;
  const int ncol  = j0 + (lane & 15);
  #pragma unroll
  for (int r = 0; r < 8; ++r) {
    sLogit[(mbase + r) * DD + ncol]        = acc0[r];
    sLogit[(mbase + r + 16) * DD + ncol]   = acc1[r];
  }
  __syncthreads();

  if (tid < MT && n0 + tid < NN) {
    const float* lrow = sLogit + tid * DD;
    float best = -1e30f; int bj = 0;
    for (int j = 0; j < DD; ++j) {
      float v = lrow[j] + bias[j];
      if (v > best) { best = v; bj = j; }
    }
    idx_out[n0 + tid] = bj;
  }
}

// ---------------------------------------------------------------------------
// Output: log_softmax( b_out + sum_l W_out[l*128 + idx_l[n], :] )
// ---------------------------------------------------------------------------
__global__ void gnn_out_kernel(const int* __restrict__ idx,       // [5][N]
                               const float* __restrict__ Wout,    // [640][40]
                               const float* __restrict__ bout,    // [40]
                               float* __restrict__ out) {
  int n = blockIdx.x * blockDim.x + threadIdx.x;
  if (n >= NN) return;
  float acc[OUTC];
  for (int j = 0; j < OUTC; ++j) acc[j] = bout[j];
  for (int l = 0; l < LLAY + 1; ++l) {
    const float* wr = Wout + ((size_t)(l * DD + idx[(size_t)l * NN + n])) * OUTC;
    for (int j = 0; j < OUTC; ++j) acc[j] += wr[j];
  }
  float m = -1e30f;
  for (int j = 0; j < OUTC; ++j) m = acc[j] > m ? acc[j] : m;
  float s = 0.0f;
  for (int j = 0; j < OUTC; ++j) s += __expf(acc[j] - m);
  float lse = m + __logf(s);
  float* orow = out + (size_t)n * OUTC;
  for (int j = 0; j < OUTC; ++j) orow[j] = acc[j] - lse;
}

// ---------------------------------------------------------------------------
extern "C" void kernel_launch(void* const* d_in, const int* in_sizes, int n_in,
                              void* d_out, int out_size, void* d_ws, size_t ws_size,
                              hipStream_t stream) {
  const float*     x        = (const float*)d_in[0];
  const long long* ei       = (const long long*)d_in[1];   // [2][E] int64
  const float*     W_in     = (const float*)d_in[2];       // [128][128]
  const float*     b_in     = (const float*)d_in[3];       // [128]
  const float*     W_layers = (const float*)d_in[4];       // [4][1408][128]
  const float*     b_layers = (const float*)d_in[5];       // [4][128]
  const float*     W_out    = (const float*)d_in[6];       // [640][40]
  const float*     b_out    = (const float*)d_in[7];       // [40]
  float* out = (float*)d_out;

  // Workspace carve (256B aligned): ~41 MB total
  char* ws = (char*)d_ws;
  size_t off = 0;
  auto carve = [&](size_t bytes) -> void* {
    void* p = ws + off;
    off = (off + bytes + 255) & ~(size_t)255;
    return p;
  };
  _Float16* xh   = (_Float16*)carve((size_t)NN * DD * sizeof(_Float16));          // 12.8 MB
  _Float16* WTin = (_Float16*)carve((size_t)DD * DD * sizeof(_Float16));          // 32 KB
  _Float16* WTl  = (_Float16*)carve((size_t)LLAY * DD * KTOT * sizeof(_Float16)); // 1.44 MB
  unsigned* cnt  = (unsigned*)carve((size_t)NN * DD * sizeof(unsigned));          // 25.6 MB
  int*      idx  = (int*)carve((size_t)(LLAY + 1) * NN * sizeof(int));            // 1 MB
  (void)ws_size; (void)in_sizes; (void)n_in; (void)out_size;

  // Precision prep: x -> f16, weights -> transposed f16
  f32_to_f16_kernel<<<(NN * DD + 255) / 256, 256, 0, stream>>>(x, xh, NN * DD);
  transpose_f16_kernel<<<(DD * DD + 255) / 256, 256, 0, stream>>>(W_in, WTin, DD, DD);
  for (int l = 0; l < LLAY; ++l)
    transpose_f16_kernel<<<(KTOT * DD + 255) / 256, 256, 0, stream>>>(
        W_layers + (size_t)l * KTOT * DD, WTl + (size_t)l * DD * KTOT, KTOT, DD);

  // Input layer -> idx[0]
  gnn_input_kernel<<<NN / 16, 256, 0, stream>>>(xh, WTin, b_in, idx);

  // 4 GNN layers
  const int layer_blocks = (NN + MT - 1) / MT;
  for (int l = 0; l < LLAY; ++l) {
    zero_u32_kernel<<<(NN * DD + 255) / 256, 256, 0, stream>>>(cnt, NN * DD);
    edge_count_kernel<<<(EE + 255) / 256, 256, 0, stream>>>(
        ei, ei + EE, idx + (size_t)l * NN, cnt);
    gnn_layer_kernel<<<layer_blocks, 256, 0, stream>>>(
        cnt, idx + (size_t)l * NN,
        WTl + (size_t)l * DD * KTOT, b_layers + (size_t)l * DD,
        idx + (size_t)(l + 1) * NN);
  }

  // Output gather + log_softmax
  gnn_out_kernel<<<(NN + 255) / 256, 256, 0, stream>>>(idx, W_out, b_out, out);
}
